// Neural_UCB_23055384445435
// MI455X (gfx1250) — matrix operand order repaired and checked by reading, not compiled, not run on hardware
//
#include <hip/hip_runtime.h>
#include <hip/hip_bf16.h>
#include <math.h>

// ---------------- constants from the reference ----------------
#define EMB    64
#define NROWS  100000
#define NCOLS  100000
#define NSEPA  17
#define ERC    1000000
#define ESC    400000
#define ESR    400000
#define ESS    289
#define EPSV   1e-5f

typedef __attribute__((ext_vector_type(16))) __bf16 v16bf;
typedef __attribute__((ext_vector_type(8)))  float  v8f;

// ---------------- bf16 helpers ----------------
static __device__ __forceinline__ unsigned short f32_to_bf16_bits(float f) {
  unsigned u = __builtin_bit_cast(unsigned, f);
  unsigned r = u + 0x7FFFu + ((u >> 16) & 1u);   // round-to-nearest-even
  return (unsigned short)(r >> 16);
}
static __device__ __forceinline__ __bf16 bits_to_bf16(unsigned short h) {
  return __builtin_bit_cast(__bf16, h);
}
static __device__ __forceinline__ __bf16 f2bf(float f) { return bits_to_bf16(f32_to_bf16_bits(f)); }
static __device__ __forceinline__ float bfbits2f(unsigned short h) {
  return __builtin_bit_cast(float, ((unsigned)h) << 16);
}

static __device__ __forceinline__ v8f v8f_zero() {
  v8f z = {0.f,0.f,0.f,0.f,0.f,0.f,0.f,0.f}; return z;
}
static __device__ __forceinline__ v8f wmma_bf16(v16bf a, v16bf b, v8f c) {
  return __builtin_amdgcn_wmma_f32_16x16x32_bf16(false, a, false, b, (short)0, c, false, false);
}

// A-fragment (16x32 bf16, MxK) from f32 row pointer (per-lane row) via b128 loads.
// Lane layout: m = lane&15, kbase = (lane>>4)*8; elems 0..7 -> K=kb..kb+7, 8..15 -> K=kb+16..kb+23
static __device__ __forceinline__ v16bf load_afrag_g(const float* __restrict__ r, int kt, int lane) {
  int kb = kt * 32 + ((lane >> 4) << 3);
  const float4* p0 = reinterpret_cast<const float4*>(r + kb);
  const float4* p1 = reinterpret_cast<const float4*>(r + kb + 16);
  float4 a0 = p0[0], a1 = p0[1], b0 = p1[0], b1 = p1[1];
  v16bf a;
  a[0] = f2bf(a0.x); a[1] = f2bf(a0.y); a[2]  = f2bf(a0.z); a[3]  = f2bf(a0.w);
  a[4] = f2bf(a1.x); a[5] = f2bf(a1.y); a[6]  = f2bf(a1.z); a[7]  = f2bf(a1.w);
  a[8] = f2bf(b0.x); a[9] = f2bf(b0.y); a[10] = f2bf(b0.z); a[11] = f2bf(b0.w);
  a[12] = f2bf(b1.x); a[13] = f2bf(b1.y); a[14] = f2bf(b1.z); a[15] = f2bf(b1.w);
  return a;
}
// Same, from an LDS bf16-bits tile [16][ld]
static __device__ __forceinline__ v16bf load_afrag_lds(const unsigned short* t, int ld, int kt, int lane) {
  int m = lane & 15;
  int kb = kt * 32 + ((lane >> 4) << 3);
  const unsigned short* r = t + m * ld;
  v16bf a;
#pragma unroll
  for (int i = 0; i < 8; ++i) a[i] = bits_to_bf16(r[kb + i]);
#pragma unroll
  for (int i = 0; i < 8; ++i) a[8 + i] = bits_to_bf16(r[kb + 16 + i]);
  return a;
}
// B-fragment from pre-packed bf16 weights: layout [(kt*4+nt)*32 + lane][16] -> two b128 loads.
static __device__ __forceinline__ v16bf load_bfrag_pk(const unsigned short* __restrict__ pk,
                                                      int kt, int nt, int lane) {
  const uint4* p = reinterpret_cast<const uint4*>(pk + (size_t)((((kt << 2) + nt) << 5) + lane) * 16);
  union { uint4 u[2]; v16bf v; } w;
  w.u[0] = p[0]; w.u[1] = p[1];
  return w.v;
}

static __device__ __forceinline__ int imin(int a, int b) { return a < b ? a : b; }

// ---------------- weight pre-pack: f32 W[K][64] -> bf16 fragment layout ----------------
// Fragment layout matches the WMMA B operand: n = lane&15 (+16*nt), K = (lane>>4)*8 + (elem&7) + (elem>>3)*16 (+32*kt)
__global__ void k_packW(const float* __restrict__ W, int krows, int ktiles,
                        unsigned short* __restrict__ out) {
  int idx = blockIdx.x * blockDim.x + threadIdx.x;
  int total = ktiles * 2048;                 // ktiles*4 frags * 32 lanes * 16 elems
  if (idx >= total) return;
  int i = idx & 15;
  int lane = (idx >> 4) & 31;
  int f = idx >> 9;
  int nt = f & 3, kt = f >> 2;
  int n = (nt << 4) + (lane & 15);
  int k = kt * 32 + ((lane >> 4) << 3) + (i & 7) + ((i >> 3) << 4);
  out[idx] = f32_to_bf16_bits(k < krows ? W[(size_t)k * 64 + n] : 0.f);
}

// ---------------- generic fill ----------------
__global__ void k_fill(float* __restrict__ p, size_t n, float v) {
  size_t i = (size_t)blockIdx.x * blockDim.x + threadIdx.x;
  size_t st = (size_t)gridDim.x * blockDim.x;
  for (; i < n; i += st) p[i] = v;
}

// ---------------- BatchNorm column stats (sum, sumsq) ----------------
template <int D>
__global__ void k_colstats(const float* __restrict__ X, int N, float* __restrict__ sums) {
  int tid = blockIdx.x * blockDim.x + threadIdx.x;
  int stride = gridDim.x * blockDim.x;
  float s[D], s2[D];
#pragma unroll
  for (int d = 0; d < D; ++d) { s[d] = 0.f; s2[d] = 0.f; }
  for (int r = tid; r < N; r += stride) {
#pragma unroll
    for (int d = 0; d < D; ++d) {
      float v = X[(size_t)r * D + d];
      s[d] += v; s2[d] += v * v;
    }
  }
#pragma unroll
  for (int d = 0; d < D; ++d) {
    atomicAdd(&sums[d], s[d]);
    atomicAdd(&sums[D + d], s2[d]);
  }
}

__global__ void k_bnfinal(const float* __restrict__ sums, int D, float invN,
                          const float* __restrict__ g, const float* __restrict__ b,
                          float* __restrict__ scale, float* __restrict__ shift) {
  int c = threadIdx.x;
  if (c < D) {
    float m = sums[c] * invN;
    float v = sums[D + c] * invN - m * m;
    float is = rsqrtf(v + EPSV);
    float sc = is * g[c];
    scale[c] = sc;
    shift[c] = b[c] - m * sc;
  }
}

// ---------------- node embedding MLP: bn -> lin(DIN->64) relu -> lin(64->64) relu ----------------
template <int DIN>
__global__ void k_embed(const float* __restrict__ X, int N,
                        const float* __restrict__ scale, const float* __restrict__ shift,
                        const unsigned short* __restrict__ W1pk, const float* __restrict__ b1,
                        const unsigned short* __restrict__ W2pk, const float* __restrict__ b2,
                        float* __restrict__ H) {
  __shared__ unsigned short sT[8][16 * 64];
  int lane = threadIdx.x & 31, wv = threadIdx.x >> 5;
  int tiles = (N + 15) >> 4;
  int tile = blockIdx.x * (blockDim.x >> 5) + wv;
  if (tile >= tiles) tile = tiles - 1;           // duplicate last tile; barrier-safe
  int row0 = tile << 4;
  int m = lane & 15, hi = lane >> 4, kb = hi * 8;
  int rowm = imin(row0 + m, N - 1);
  const float* xr = X + (size_t)rowm * DIN;

  v16bf a;
#pragma unroll
  for (int i = 0; i < 8; ++i) {
    int c = kb + i;
    a[i] = f2bf(c < DIN ? xr[c] * scale[c] + shift[c] : 0.f);
  }
#pragma unroll
  for (int i = 0; i < 8; ++i) {
    int c = kb + 16 + i;
    a[8 + i] = f2bf(c < DIN ? xr[c] * scale[c] + shift[c] : 0.f);
  }
  // layer 1: K padded to 32, one k-tile, 4 n-tiles
#pragma unroll
  for (int nt = 0; nt < 4; ++nt) {
    v8f acc = v8f_zero();
    acc = wmma_bf16(a, load_bfrag_pk(W1pk, 0, nt, lane), acc);
    int n = nt * 16 + m;
    float bias = b1[n];
#pragma unroll
    for (int j = 0; j < 8; ++j) {
      float v = fmaxf(acc[j] + bias, 0.f);
      sT[wv][(j + hi * 8) * 64 + n] = f32_to_bf16_bits(v);
    }
  }
  __syncthreads();
  // layer 2: 64x64
  v8f acc2[4];
#pragma unroll
  for (int nt = 0; nt < 4; ++nt) acc2[nt] = v8f_zero();
#pragma unroll
  for (int kt = 0; kt < 2; ++kt) {
    v16bf a2 = load_afrag_lds(sT[wv], 64, kt, lane);
#pragma unroll
    for (int nt = 0; nt < 4; ++nt)
      acc2[nt] = wmma_bf16(a2, load_bfrag_pk(W2pk, kt, nt, lane), acc2[nt]);
  }
#pragma unroll
  for (int nt = 0; nt < 4; ++nt) {
    int n = nt * 16 + m;
    float bias = b2[n];
#pragma unroll
    for (int j = 0; j < 8; ++j) {
      int grow = row0 + j + hi * 8;
      if (grow < N) H[(size_t)grow * 64 + n] = fmaxf(acc2[nt][j] + bias, 0.f);
    }
  }
}

// sepa embedding: LN over dim-1 collapses to bias; MLP is node-independent
__global__ void k_embed_sepa(const float* __restrict__ lnb1,
                             const float* __restrict__ W1, const float* __restrict__ b1,
                             const float* __restrict__ W2, const float* __restrict__ b2,
                             float* __restrict__ H) {
  __shared__ float h1[64];
  int c = threadIdx.x;
  float y0 = lnb1[0];
  h1[c] = fmaxf(y0 * W1[c] + b1[c], 0.f);
  __syncthreads();
  float s = b2[c];
  for (int j = 0; j < 64; ++j) s += h1[j] * W2[j * 64 + c];
  s = fmaxf(s, 0.f);
  for (int n = 0; n < NSEPA; ++n) H[(size_t)n * 64 + c] = s;
}

// ---------------- Y[N,64] = act(X[N,64] @ W + b); optional bf16 output ----------------
template <bool BIAS, bool RELU, bool BF16OUT>
__global__ void k_gemm64(const float* __restrict__ X, int N,
                         const unsigned short* __restrict__ Wpk, const float* __restrict__ b,
                         void* __restrict__ Yv) {
  int lane = threadIdx.x & 31, wv = threadIdx.x >> 5;
  int tiles = (N + 15) >> 4;
  int tile = blockIdx.x * (blockDim.x >> 5) + wv;
  if (tile >= tiles) return;
  int row0 = tile << 4;
  int m = lane & 15, hi = lane >> 4;
  int rm = imin(row0 + m, N - 1);
  const float* xr = X + (size_t)rm * 64;
  v8f acc[4];
#pragma unroll
  for (int nt = 0; nt < 4; ++nt) acc[nt] = v8f_zero();
#pragma unroll
  for (int kt = 0; kt < 2; ++kt) {
    v16bf a = load_afrag_g(xr, kt, lane);
#pragma unroll
    for (int nt = 0; nt < 4; ++nt)
      acc[nt] = wmma_bf16(a, load_bfrag_pk(Wpk, kt, nt, lane), acc[nt]);
  }
#pragma unroll
  for (int nt = 0; nt < 4; ++nt) {
    int n = nt * 16 + m;
    float bias = BIAS ? b[n] : 0.f;
#pragma unroll
    for (int j = 0; j < 8; ++j) {
      int grow = row0 + j + hi * 8;
      if (grow < N) {
        float u = acc[nt][j] + bias;
        if (RELU) u = fmaxf(u, 0.f);
        if (BF16OUT) ((unsigned short*)Yv)[(size_t)grow * 64 + n] = f32_to_bf16_bits(u);
        else         ((float*)Yv)[(size_t)grow * 64 + n] = u;
      }
    }
  }
}

// ---------------- bgc edge phase: msg = final_lin(relu(LN(P[tgt]+Q[src]+e@We))) + b, agg[tgt]+=msg ----------------
// P, Q stored bf16 to halve the dominant gather traffic.
template <int EDIM>
__global__ void k_edge_msg(const int* __restrict__ src, const int* __restrict__ tgt,
                           const float* __restrict__ ev, int E,
                           const unsigned short* __restrict__ P, const unsigned short* __restrict__ Q,
                           const float* __restrict__ escale, const float* __restrict__ eshift,
                           const float* __restrict__ We,
                           const float* __restrict__ lng, const float* __restrict__ lnb,
                           const unsigned short* __restrict__ Wfpk, const float* __restrict__ bf,
                           float* __restrict__ agg) {
  int lane = threadIdx.x & 31, wv = threadIdx.x >> 5;
  int tiles = (E + 15) >> 4;
  int tile = blockIdx.x * (blockDim.x >> 5) + wv;
  if (tile >= tiles) return;
  int e0 = tile << 4;
  int m = lane & 15, hi = lane >> 4;
  int em = imin(e0 + m, E - 1);
  int s = src[em], t = tgt[em];
  __builtin_prefetch(&ev[(size_t)em * EDIM], 0, 0);   // global_prefetch_b8
  float en[EDIM];
#pragma unroll
  for (int d = 0; d < EDIM; ++d)
    en[d] = ev[(size_t)em * EDIM + d] * escale[d] + eshift[d];
  const unsigned short* Pr = P + (size_t)t * 64;
  const unsigned short* Qr = Q + (size_t)s * 64;

  auto hv = [&](int c) {
    float h = bfbits2f(Pr[c]) + bfbits2f(Qr[c]);
#pragma unroll
    for (int d = 0; d < EDIM; ++d) h += en[d] * We[d * 64 + c];
    return h;
  };

  // LayerNorm stats on lanes 0..15 (one row each), broadcast to partner lanes
  float mean = 0.f, istd = 0.f;
  if (hi == 0) {
    float s1 = 0.f, s2 = 0.f;
    for (int c = 0; c < 64; ++c) { float h = hv(c); s1 += h; s2 += h * h; }
    mean = s1 * (1.f / 64.f);
    float var = s2 * (1.f / 64.f) - mean * mean;
    istd = rsqrtf(var + EPSV);
  }
  mean = __shfl(mean, m, 32);
  istd = __shfl(istd, m, 32);

  v8f acc[4];
#pragma unroll
  for (int nt = 0; nt < 4; ++nt) acc[nt] = v8f_zero();
  int kb = hi * 8;
#pragma unroll
  for (int kt = 0; kt < 2; ++kt) {
    v16bf a;
#pragma unroll
    for (int i = 0; i < 8; ++i) {
      int c = kt * 32 + kb + i;
      a[i] = f2bf(fmaxf((hv(c) - mean) * istd * lng[c] + lnb[c], 0.f));
    }
#pragma unroll
    for (int i = 0; i < 8; ++i) {
      int c = kt * 32 + kb + 16 + i;
      a[8 + i] = f2bf(fmaxf((hv(c) - mean) * istd * lng[c] + lnb[c], 0.f));
    }
#pragma unroll
    for (int nt = 0; nt < 4; ++nt)
      acc[nt] = wmma_bf16(a, load_bfrag_pk(Wfpk, kt, nt, lane), acc[nt]);
  }
  // scatter with per-edge bias
#pragma unroll
  for (int j = 0; j < 8; ++j) {
    int eg = e0 + j + hi * 8;
    if (eg < E) {
      int te = tgt[eg];
#pragma unroll
      for (int nt = 0; nt < 4; ++nt) {
        int n = nt * 16 + m;
        atomicAdd(&agg[(size_t)te * 64 + n], acc[nt][j] + bf[n]);
      }
    }
  }
}

// ---------------- bgc node update: newR = out2( relu(out1( [LN(agg), R] )) ) ----------------
__global__ void k_node_update(const float* __restrict__ agg, const float* __restrict__ R, int N,
                              const float* __restrict__ lng, const float* __restrict__ lnb,
                              const unsigned short* __restrict__ W1pk, const float* __restrict__ b1,
                              const unsigned short* __restrict__ W2pk, const float* __restrict__ b2,
                              float* __restrict__ Y) {
  __shared__ unsigned short sA[8][16 * 128];
  __shared__ unsigned short sB[8][16 * 64];
  int lane = threadIdx.x & 31, wv = threadIdx.x >> 5;
  int tiles = (N + 15) >> 4;
  int tile = blockIdx.x * (blockDim.x >> 5) + wv;
  if (tile >= tiles) tile = tiles - 1;
  int row0 = tile << 4;
  int m = lane & 15, hi = lane >> 4;

  if (lane < 16) {
    int r = imin(row0 + lane, N - 1);
    const float* ar = agg + (size_t)r * 64;
    float s1 = 0.f, s2 = 0.f;
    for (int c = 0; c < 64; ++c) { float v = ar[c]; s1 += v; s2 += v * v; }
    float mean = s1 * (1.f / 64.f);
    float var = s2 * (1.f / 64.f) - mean * mean;
    float istd = rsqrtf(var + EPSV);
    unsigned short* dst = &sA[wv][lane * 128];
    for (int c = 0; c < 64; ++c)
      dst[c] = f32_to_bf16_bits((ar[c] - mean) * istd * lng[c] + lnb[c]);
    const float* rr = R + (size_t)r * 64;
    for (int c = 0; c < 64; ++c) dst[64 + c] = f32_to_bf16_bits(rr[c]);
  }
  __syncthreads();

  v8f acc[4];
#pragma unroll
  for (int nt = 0; nt < 4; ++nt) acc[nt] = v8f_zero();
#pragma unroll
  for (int kt = 0; kt < 4; ++kt) {
    v16bf a = load_afrag_lds(sA[wv], 128, kt, lane);
#pragma unroll
    for (int nt = 0; nt < 4; ++nt)
      acc[nt] = wmma_bf16(a, load_bfrag_pk(W1pk, kt, nt, lane), acc[nt]);
  }
#pragma unroll
  for (int nt = 0; nt < 4; ++nt) {
    int n = nt * 16 + m;
    float bias = b1[n];
#pragma unroll
    for (int j = 0; j < 8; ++j)
      sB[wv][(j + hi * 8) * 64 + n] = f32_to_bf16_bits(fmaxf(acc[nt][j] + bias, 0.f));
  }
  __syncthreads();

  v8f acc2[4];
#pragma unroll
  for (int nt = 0; nt < 4; ++nt) acc2[nt] = v8f_zero();
#pragma unroll
  for (int kt = 0; kt < 2; ++kt) {
    v16bf a = load_afrag_lds(sB[wv], 64, kt, lane);
#pragma unroll
    for (int nt = 0; nt < 4; ++nt)
      acc2[nt] = wmma_bf16(a, load_bfrag_pk(W2pk, kt, nt, lane), acc2[nt]);
  }
#pragma unroll
  for (int nt = 0; nt < 4; ++nt) {
    int n = nt * 16 + m;
    float bias = b2[n];
#pragma unroll
    for (int j = 0; j < 8; ++j) {
      int grow = row0 + j + hi * 8;
      if (grow < N) Y[(size_t)grow * 64 + n] = acc2[nt][j] + bias;   // no outer relu
    }
  }
}

// ---------------- tiny dense helpers (17-node transformer etc.) ----------------
__global__ void k_small_lin(const float* __restrict__ X, int M,
                            const float* __restrict__ W, const float* __restrict__ b,
                            float* __restrict__ Y) {
  for (int idx = threadIdx.x; idx < M * 64; idx += blockDim.x) {
    int n = idx >> 6, c = idx & 63;
    float s = b[c];
    for (int j = 0; j < 64; ++j) s += X[(size_t)n * 64 + j] * W[j * 64 + c];
    Y[idx] = s;
  }
}

__global__ void k_tconv(const int* __restrict__ src, const int* __restrict__ tgt,
                        const float* __restrict__ ev, const float* __restrict__ We,
                        const float* __restrict__ q, const float* __restrict__ k,
                        const float* __restrict__ v, const float* __restrict__ skip,
                        float* __restrict__ out) {
  __shared__ float aA[ESS * 4];
  __shared__ float mx[NSEPA * 4];
  __shared__ float dn[NSEPA * 4];
  for (int idx = threadIdx.x; idx < ESS * 4; idx += blockDim.x) {
    int e = idx >> 2, h = idx & 3;
    int s = src[e], t = tgt[e];
    float acc = 0.f;
    for (int d = 0; d < 16; ++d) {
      int c = h * 16 + d;
      acc += q[(size_t)t * 64 + c] * (k[(size_t)s * 64 + c] + ev[e] * We[c]);
    }
    aA[idx] = acc * 0.25f;   // 1/sqrt(16)
  }
  __syncthreads();
  for (int idx = threadIdx.x; idx < NSEPA * 4; idx += blockDim.x) {
    int n = idx >> 2, h = idx & 3;
    float mm = -3.4e38f;
    for (int e = 0; e < ESS; ++e)
      if (tgt[e] == n) mm = fmaxf(mm, aA[e * 4 + h]);
    float s = 0.f;
    for (int e = 0; e < ESS; ++e)
      if (tgt[e] == n) s += expf(aA[e * 4 + h] - mm);
    mx[idx] = mm; dn[idx] = s;
  }
  __syncthreads();
  for (int idx = threadIdx.x; idx < NSEPA * 64; idx += blockDim.x) {
    int n = idx >> 6, c = idx & 63, h = c >> 4;
    float d = dn[n * 4 + h], mm = mx[n * 4 + h];
    float o = 0.f;
    for (int e = 0; e < ESS; ++e) {
      if (tgt[e] == n) {
        float w = expf(aA[e * 4 + h] - mm);
        o += w * (v[(size_t)src[e] * 64 + c] + ev[e] * We[c]);
      }
    }
    o = (d > 0.f) ? o / d : 0.f;
    out[idx] = o + skip[idx];
  }
}

__global__ void k_sepa_out(const float* __restrict__ sepa, const float* __restrict__ satt,
                           const float* __restrict__ xs,
                           const float* __restrict__ W, const float* __restrict__ b,
                           float* __restrict__ Y) {
  for (int idx = threadIdx.x; idx < NSEPA * 64; idx += blockDim.x) {
    int n = idx >> 6, c = idx & 63;
    float s = b[c];
    for (int j = 0; j < 64; ++j) s += sepa[(size_t)n * 64 + j] * W[j * 64 + c];
    for (int j = 0; j < 64; ++j) s += satt[(size_t)n * 64 + j] * W[(64 + j) * 64 + c];
    s += xs[n] * W[128 * 64 + c];
    Y[idx] = fmaxf(s, 0.f);
  }
}

__global__ void k_colmean(const float* __restrict__ X, int N, float invN, float* __restrict__ out) {
  int c = threadIdx.x;                      // 64 threads
  int chunk = (N + gridDim.x - 1) / gridDim.x;
  int r0 = blockIdx.x * chunk;
  int r1 = imin(N, r0 + chunk);
  float s = 0.f;
  for (int r = r0; r < r1; ++r) s += X[(size_t)r * 64 + c];
  atomicAdd(&out[c], s * invN);
}

__global__ void k_final(const float* __restrict__ pooled,
                        const float* __restrict__ W1, const float* __restrict__ b1,
                        const float* __restrict__ W2, const float* __restrict__ b2,
                        float* __restrict__ out) {
  __shared__ float red[64];
  int j = threadIdx.x;                      // 64 threads
  float t = b1[j];
  for (int i = 0; i < 192; ++i) t += pooled[i] * W1[i * 64 + j];
  t = fmaxf(t, 0.f);
  red[j] = t * W2[j];
  __syncthreads();
  if (j == 0) {
    float s = 0.f;
    for (int i = 0; i < 64; ++i) s += red[i];
    s += b2[0];
    out[0] = 1.f / (1.f + expf(-s));
  }
}

// ---------------- host orchestration ----------------
static inline int tiles16(int n) { return (n + 15) / 16; }
static inline int wblocks(int n) { return (tiles16(n) + 7) / 8; }   // 8 waves / 256-thread block

extern "C" void kernel_launch(void* const* d_in, const int* in_sizes, int n_in,
                              void* d_out, int out_size, void* d_ws, size_t ws_size,
                              hipStream_t stream) {
  (void)in_sizes; (void)n_in; (void)out_size; (void)ws_size;
  auto F = [&](int i) { return (const float*)d_in[i]; };
  auto I = [&](int i) { return (const int*)d_in[i]; };

  // ---- param leaf indices (jax pytree: recursively sorted dict keys), base 11 ----
  // bgc block layout (+14 leaves): 0 edge.W, 1 fl.W, 2 fl.b, 3 fln.b, 4 fln.g,
  //   5 left.W, 6 left.b, 7 out1.W, 8 out1.b, 9 out2.W, 10 out2.b, 11 pln.b, 12 pln.g, 13 right.W
  const int P_C2R = 11, P_C2S = 25;
  const int P_COLBN_B = 39, P_COLBN_G = 40, P_COLL1_W = 41, P_COLL1_B = 42, P_COLL2_W = 43, P_COLL2_B = 44;
  const int P_ENRC_B = 45, P_ENRC_G = 46, P_ENR_B = 47, P_ENR_G = 48, P_ENS_B = 49, P_ENS_G = 50;
  const int P_OUT1_W = 51, P_OUT1_B = 52, P_OUT2_W = 53, P_OUT2_B = 54;
  const int P_R2C = 55, P_R2S = 69;
  const int P_ROWBN_B = 83, P_ROWBN_G = 84, P_ROWL1_W = 85, P_ROWL1_B = 86, P_ROWL2_W = 87, P_ROWL2_B = 88;
  const int P_ROWOUT_W = 89, P_ROWOUT_B = 90;
  const int P_S2R = 91;
  const int P_SEPAL1_W = 105, P_SEPAL1_B = 106, P_SEPAL2_W = 107, P_SEPAL2_B = 108;
  const int P_SEPALN_B = 109;
  const int P_SEPAOUT_W = 111, P_SEPAOUT_B = 112;
  const int P_TF_EW = 113, P_TF_KW = 114, P_TF_KB = 115, P_TF_QW = 116, P_TF_QB = 117,
            P_TF_SW = 118, P_TF_SB = 119, P_TF_VW = 120, P_TF_VB = 121;

  // ---- workspace layout (floats): 6 big buffers + small region + packed weights ----
  float* wsf = (float*)d_ws;
  const size_t SZ = (size_t)NROWS * EMB;
  float* rowA = wsf + 0 * SZ;
  float* rowB = wsf + 1 * SZ;
  float* colA = wsf + 2 * SZ;
  float* colB = wsf + 3 * SZ;
  float* Qb   = wsf + 4 * SZ;
  float* agg  = wsf + 5 * SZ;
  float* sp   = wsf + 6 * SZ;
  float* rowSum = sp + 0;    float* colSum = sp + 40;
  float* ercSum = sp + 80;   float* esrSum = sp + 88;   float* escSum = sp + 92;
  float* rowSc = sp + 96;    float* rowSh = sp + 128;
  float* colSc = sp + 160;   float* colSh = sp + 192;
  float* ercSc = sp + 224;   float* ercSh = sp + 228;
  float* esrSc = sp + 232;   float* esrSh = sp + 236;
  float* escSc = sp + 240;   float* escSh = sp + 244;
  float* sepaA = sp + 256;   float* sepaB = sp + 1600;
  float* tq = sp + 2944;     float* tk = sp + 4288;
  float* tv = sp + 5632;     float* tskip = sp + 6976;
  float* satt = sp + 8320;   float* satt2 = sp + 9664;
  float* pooled = sp + 11008;
  // packed bf16 weights (ushort units), 16-byte aligned
  unsigned short* pk = (unsigned short*)(sp + 16384);
  unsigned short* pkWl = pk + 0;            // 4096 (64x64)
  unsigned short* pkWr = pk + 4096;         // 4096
  unsigned short* pkWf = pk + 8192;         // 4096
  unsigned short* pkW1 = pk + 12288;        // 8192 (128x64)
  unsigned short* pkW2 = pk + 20480;        // 4096
  unsigned short* pkRow1 = pk + 24576;      // 2048 (K padded to 32)
  unsigned short* pkRow2 = pk + 26624;      // 4096
  unsigned short* pkCol1 = pk + 30720;      // 2048
  unsigned short* pkCol2 = pk + 32768;      // 4096
  unsigned short* pkRout = pk + 36864;      // 4096

  auto packW = [&](const float* W, int krows, int ktiles, unsigned short* dst) {
    int tot = ktiles * 2048;
    k_packW<<<(tot + 255) / 256, 256, 0, stream>>>(W, krows, ktiles, dst);
  };

  // ---- BN statistics ----
  k_fill<<<1, 256, 0, stream>>>(sp, (size_t)96, 0.f);
  k_fill<<<1, 256, 0, stream>>>(pooled, (size_t)192, 0.f);
  k_colstats<18><<<256, 256, 0, stream>>>(F(0), NROWS, rowSum);
  k_colstats<19><<<256, 256, 0, stream>>>(F(1), NCOLS, colSum);
  k_colstats<2><<<256, 256, 0, stream>>>(F(4), ERC, ercSum);
  k_colstats<1><<<256, 256, 0, stream>>>(F(8), ESR, esrSum);
  k_colstats<1><<<256, 256, 0, stream>>>(F(6), ESC, escSum);
  k_bnfinal<<<1, 32, 0, stream>>>(rowSum, 18, 1.f / NROWS, F(P_ROWBN_G), F(P_ROWBN_B), rowSc, rowSh);
  k_bnfinal<<<1, 32, 0, stream>>>(colSum, 19, 1.f / NCOLS, F(P_COLBN_G), F(P_COLBN_B), colSc, colSh);
  k_bnfinal<<<1, 32, 0, stream>>>(ercSum, 2, 1.f / ERC, F(P_ENRC_G), F(P_ENRC_B), ercSc, ercSh);
  k_bnfinal<<<1, 32, 0, stream>>>(esrSum, 1, 1.f / ESR, F(P_ENR_G), F(P_ENR_B), esrSc, esrSh);
  k_bnfinal<<<1, 32, 0, stream>>>(escSum, 1, 1.f / ESC, F(P_ENS_G), F(P_ENS_B), escSc, escSh);

  // ---- node embeddings ----
  packW(F(P_ROWL1_W), 18, 1, pkRow1);
  packW(F(P_ROWL2_W), 64, 2, pkRow2);
  packW(F(P_COLL1_W), 19, 1, pkCol1);
  packW(F(P_COLL2_W), 64, 2, pkCol2);
  k_embed<18><<<wblocks(NROWS), 256, 0, stream>>>(F(0), NROWS, rowSc, rowSh,
      pkRow1, F(P_ROWL1_B), pkRow2, F(P_ROWL2_B), rowA);
  k_embed<19><<<wblocks(NCOLS), 256, 0, stream>>>(F(1), NCOLS, colSc, colSh,
      pkCol1, F(P_COLL1_B), pkCol2, F(P_COLL2_B), colA);
  k_embed_sepa<<<1, 64, 0, stream>>>(F(P_SEPALN_B), F(P_SEPAL1_W), F(P_SEPAL1_B),
                                     F(P_SEPAL2_W), F(P_SEPAL2_B), sepaA);

  // ---- bgc driver (bf16 P aliases newR: P is dead before node_update writes newR) ----
  auto run_bgc = [&](const float* left, int Nl, const float* right, int Nr,
                     const int* es, const int* et, const float* ev, int E, int EDIM,
                     const float* eSc, const float* eSh, int pb, float* newR) {
    packW(F(pb + 5), 64, 2, pkWl);     // left lin
    packW(F(pb + 13), 64, 2, pkWr);    // right lin
    packW(F(pb + 1), 64, 2, pkWf);     // final_lin
    packW(F(pb + 7), 128, 4, pkW1);    // out1
    packW(F(pb + 9), 64, 2, pkW2);     // out2
    unsigned short* Pb16 = (unsigned short*)newR;
    unsigned short* Qb16 = (unsigned short*)Qb;
    k_gemm64<true, false, true><<<wblocks(Nr), 256, 0, stream>>>(right, Nr, pkWl, F(pb + 6), Pb16);
    k_gemm64<false, false, true><<<wblocks(Nl), 256, 0, stream>>>(left, Nl, pkWr, nullptr, Qb16);
    size_t an = (size_t)Nr * EMB;
    k_fill<<<(int)((an + 65535) / 65536) + 1, 256, 0, stream>>>(agg, an, 0.f);
    int eb = (((E + 15) / 16) + 7) / 8;
    if (EDIM == 2)
      k_edge_msg<2><<<eb, 256, 0, stream>>>(es, et, ev, E, Pb16, Qb16, eSc, eSh,
          F(pb + 0), F(pb + 4), F(pb + 3), pkWf, F(pb + 2), agg);
    else
      k_edge_msg<1><<<eb, 256, 0, stream>>>(es, et, ev, E, Pb16, Qb16, eSc, eSh,
          F(pb + 0), F(pb + 4), F(pb + 3), pkWf, F(pb + 2), agg);
    k_node_update<<<wblocks(Nr), 256, 0, stream>>>(agg, right, Nr,
        F(pb + 12), F(pb + 11), pkW1, F(pb + 8), pkW2, F(pb + 10), newR);
  };

  const int* rc0 = I(3);            const int* rc1 = I(3) + ERC;
  const int* sc0 = I(5);            const int* sc1 = I(5) + ESC;
  const int* sr0 = I(7);            const int* sr1 = I(7) + ESR;

  run_bgc(colA, NCOLS, rowA, NROWS, rc1, rc0, F(4), ERC, 2, ercSc, ercSh, P_C2R, rowB); // cols->rows
  run_bgc(rowB, NROWS, colA, NCOLS, rc0, rc1, F(4), ERC, 2, ercSc, ercSh, P_R2C, colB); // rows->cols
  run_bgc(colB, NCOLS, sepaA, NSEPA, sc1, sc0, F(6), ESC, 1, escSc, escSh, P_C2S, sepaB); // cols->sepas
  run_bgc(sepaB, NSEPA, rowB, NROWS, sr0, sr1, F(8), ESR, 1, esrSc, esrSh, P_S2R, rowA); // sepas->rows
  run_bgc(rowA, NROWS, sepaB, NSEPA, sr1, sr0, F(8), ESR, 1, esrSc, esrSh, P_R2S, sepaA); // rows->sepas

  // ---- 17-node TransformerConv ----
  k_small_lin<<<1, 256, 0, stream>>>(sepaA, NSEPA, F(P_TF_QW), F(P_TF_QB), tq);
  k_small_lin<<<1, 256, 0, stream>>>(sepaA, NSEPA, F(P_TF_KW), F(P_TF_KB), tk);
  k_small_lin<<<1, 256, 0, stream>>>(sepaA, NSEPA, F(P_TF_VW), F(P_TF_VB), tv);
  k_small_lin<<<1, 256, 0, stream>>>(sepaA, NSEPA, F(P_TF_SW), F(P_TF_SB), tskip);
  k_tconv<<<1, 256, 0, stream>>>(I(9), I(9) + ESS, F(10), F(P_TF_EW), tq, tk, tv, tskip, satt);
  k_sepa_out<<<1, 256, 0, stream>>>(sepaA, satt, F(2), F(P_SEPAOUT_W), F(P_SEPAOUT_B), satt2);

  // ---- row head + pooling + final ----
  packW(F(P_ROWOUT_W), 64, 2, pkRout);
  k_gemm64<true, true, false><<<wblocks(NROWS), 256, 0, stream>>>(rowA, NROWS, pkRout, F(P_ROWOUT_B), Qb);
  k_colmean<<<1, 64, 0, stream>>>(satt2, NSEPA, 1.f / NSEPA, pooled);
  k_colmean<<<128, 64, 0, stream>>>(Qb, NROWS, 1.f / NROWS, pooled + 64);
  k_colmean<<<128, 64, 0, stream>>>(colB, NCOLS, 1.f / NCOLS, pooled + 128);
  k_final<<<1, 64, 0, stream>>>(pooled, F(P_OUT1_W), F(P_OUT1_B), F(P_OUT2_W), F(P_OUT2_B), (float*)d_out);
}
